// GraphAttentionLayer_15195594293946
// MI455X (gfx1250) — matrix operand order repaired and checked by reading, not compiled, not run on hardware
//
#include <hip/hip_runtime.h>
#include <hip/hip_bf16.h>
#include <cstdint>
#include <cstddef>

// GAT layer for MI455X (gfx1250).
// Roofline: adj stream = 256MB -> ~11us floor @ 23.3TB/s (non-temporal, so it
// does not evict the L2-resident reuse set: WhT 4MB, Wh2 32KB; L2=192MB).
// 32-row blocks keep WhT L2 re-read traffic at 1GB (vs 2GB with 16-row blocks)
// so the B-fragment path stays under the adj HBM floor.
// Attention matmul (34.4 GFLOP) via v_wmma_f32_16x16x32_bf16 with f32
// accumulation; the small exact X@W GEMM via v_wmma_f32_16x16x4_f32.
// Fused one-pass softmax with a global upper-bound shift (softmax is
// shift-invariant per row; all p<=1), so no online-max rescaling is needed.

#define N_NODES 8192
#define D_IN    512
#define D_OUT   256
#define JT      64          // j-tile (K per iteration of the attention loop)
#define ROWS    32          // rows per block in the fused kernel
#define PSTR    72          // padded LDS row stride (ushorts) for the P tile

typedef __attribute__((ext_vector_type(2)))  float          v2f;
typedef __attribute__((ext_vector_type(8)))  float          v8f;
typedef __attribute__((ext_vector_type(16))) __bf16         v16bf;
typedef __attribute__((ext_vector_type(4)))  unsigned int   u32x4;
typedef __attribute__((ext_vector_type(4)))  int            i32x4;
typedef __attribute__((ext_vector_type(4)))  float          f32x4;
typedef __attribute__((ext_vector_type(4)))  unsigned short u16x4;

__device__ __forceinline__ unsigned short f2bf(float x) {
  unsigned int u = __float_as_uint(x);
  u += 0x7FFFu + ((u >> 16) & 1u);          // round-to-nearest-even
  return (unsigned short)(u >> 16);
}

// ---------------------------------------------------------------------------
// Kernel 1: Wh = X @ W  (f32, exact) — 16x16 tile per wave, K-step 4.
// A 16x4: lane L(0-15): row=L, K={k,k+1}; lane L(16-31): row=L-16, K={k+2,k+3}.
// B 4x16 mirrors A over N. C/D: VGPR r -> row r (lanes 0-15) / r+8 (lanes 16-31).
// ---------------------------------------------------------------------------
__global__ void __launch_bounds__(128)
k_wh_gemm(const float* __restrict__ X, const float* __restrict__ W,
          float* __restrict__ Wh) {
  const int lane = threadIdx.x & 31;
  const int wave = threadIdx.x >> 5;
  const int r0   = blockIdx.x * 16;
  const int cc   = blockIdx.y * 64 + wave * 16 + (lane & 15);
  const int ar   = r0 + (lane & 15);
  const int ko   = (lane < 16) ? 0 : 2;
  v8f acc = {};
  const float* xrow = X + (size_t)ar * D_IN + ko;
  for (int k = 0; k < D_IN; k += 4) {
    v2f a = *(const v2f*)(xrow + k);
    v2f b;
    b.x = W[(size_t)(k + ko)     * D_OUT + cc];
    b.y = W[(size_t)(k + ko + 1) * D_OUT + cc];
    acc = __builtin_amdgcn_wmma_f32_16x16x4_f32(false, a, false, b,
                                                (short)0, acc, false, false);
  }
  const int rofs = (lane < 16) ? 0 : 8;
  for (int r = 0; r < 8; ++r)
    Wh[(size_t)(r0 + r + rofs) * D_OUT + cc] = acc[r];
}

// ---------------------------------------------------------------------------
// Kernel 2a: WhT[c][r] = bf16(Wh[r][c])  (tiled transpose, coalesced both ways)
// ---------------------------------------------------------------------------
__global__ void __launch_bounds__(256)
k_transpose_bf16(const float* __restrict__ Wh, unsigned short* __restrict__ WhT) {
  __shared__ unsigned short t[32][33];
  const int c0 = blockIdx.x * 32;    // over D_OUT (8 tiles)
  const int r0 = blockIdx.y * 32;    // over N    (256 tiles)
  const int tx = threadIdx.x;        // 0..31
  const int ty = threadIdx.y;        // 0..7
  for (int k = 0; k < 32; k += 8)
    t[ty + k][tx] = f2bf(Wh[(size_t)(r0 + ty + k) * D_OUT + c0 + tx]);
  __syncthreads();
  for (int k = 0; k < 32; k += 8)
    WhT[(size_t)(c0 + ty + k) * N_NODES + r0 + tx] = t[tx][ty + k];
}

// ---------------------------------------------------------------------------
// Kernel 2b: Wh1/Wh2 = Wh @ a[:256] / Wh @ a[256:]  (one wave per row)
// ---------------------------------------------------------------------------
__global__ void __launch_bounds__(256)
k_wh12(const float* __restrict__ Wh, const float* __restrict__ a,
       float* __restrict__ Wh1, float* __restrict__ Wh2) {
  const int row  = blockIdx.x * 8 + (threadIdx.x >> 5);
  const int lane = threadIdx.x & 31;
  float s1 = 0.f, s2 = 0.f;
  for (int c = lane; c < D_OUT; c += 32) {
    float v = Wh[(size_t)row * D_OUT + c];
    s1 += v * a[c];
    s2 += v * a[D_OUT + c];
  }
  for (int o = 16; o > 0; o >>= 1) {
    s1 += __shfl_down(s1, o, 32);
    s2 += __shfl_down(s2, o, 32);
  }
  if (lane == 0) { Wh1[row] = s1; Wh2[row] = s2; }
}

// ---------------------------------------------------------------------------
// Kernel 2c: shift = leakyrelu(max(Wh1) + max(Wh2))  (global softmax shift)
// ---------------------------------------------------------------------------
__global__ void __launch_bounds__(256)
k_shift(const float* __restrict__ Wh1, const float* __restrict__ Wh2,
        float* __restrict__ sh) {
  __shared__ float m1s[256], m2s[256];
  float m1 = -1e30f, m2 = -1e30f;
  for (int i = threadIdx.x; i < N_NODES; i += 256) {
    m1 = fmaxf(m1, Wh1[i]);
    m2 = fmaxf(m2, Wh2[i]);
  }
  m1s[threadIdx.x] = m1; m2s[threadIdx.x] = m2;
  __syncthreads();
  for (int o = 128; o > 0; o >>= 1) {
    if (threadIdx.x < o) {
      m1s[threadIdx.x] = fmaxf(m1s[threadIdx.x], m1s[threadIdx.x + o]);
      m2s[threadIdx.x] = fmaxf(m2s[threadIdx.x], m2s[threadIdx.x + o]);
    }
    __syncthreads();
  }
  if (threadIdx.x == 0) {
    float s = m1s[0] + m2s[0];
    *sh = (s > 0.f) ? s : 0.2f * s;
  }
}

// ---------------------------------------------------------------------------
// Kernel 3: fused mask->exp->(P@Wh) with final /l_i normalization.
// Block = 32 rows, 512 threads = 16 waves; wave w owns output cols 16w..16w+15.
// j-tiles of 64: 512 threads compute scores (adj loads non-temporal + fully
// coalesced: 16 thr/row -> 256B contiguous), write a double-buffered bf16
// P tile (32x64) to LDS (stride 72 -> conflict-free ds_load_b128 A-frags).
// Each wave then runs 2 K-subtiles x 2 row-groups = 4 v_wmma_f32_16x16x32_bf16,
// with each B fragment loaded once and reused by both row-groups.
// ONE barrier per tile thanks to double buffering (__syncthreads waits DScnt,
// so reads of buffer b in iter n complete before barrier n+1; the next write
// to b happens after barrier n+1).
// ---------------------------------------------------------------------------
__global__ void __launch_bounds__(512)
k_gat_attn(const int* __restrict__ adj, const float* __restrict__ Wh1,
           const float* __restrict__ Wh2, const unsigned short* __restrict__ WhT,
           const float* __restrict__ shp, float* __restrict__ out) {
  __shared__ unsigned short Pt[2][ROWS * PSTR];  // 2 x (32 rows x 64 cols bf16)
  __shared__ float lpart[ROWS * 16];
  __shared__ float lrow[ROWS];

  const int tid  = threadIdx.x;
  const int lane = tid & 31;
  const int wave = tid >> 5;                    // 0..15
  const int row0 = blockIdx.x * ROWS;
  const float shift = *shp;

  // score-producer mapping: 16 threads per row, 4 cols each
  const int i  = tid >> 4;                      // 0..31 (row within block)
  const int jb = (tid & 15) * 4;                // 0,4,...,60 (col within tile)
  const float wh1 = Wh1[row0 + i];

  // WMMA fragment mapping (ISA 7.12.2 layouts)
  const int arow    = lane & 15;
  const int kbA     = (lane < 16) ? 0 : 8;      // A: K base within 16-group
  const int kbB     = (lane < 16) ? 0 : 8;      // B: K base within 16-group
  const int colbase = wave * 16 + (lane & 15);  // this wave's output column

  float rsum = 0.f;
  v8f acc0 = {}, acc1 = {};                     // rows 0-15 / rows 16-31
  union Frag { u32x4 q[2]; v16bf v; };

  const int* adjrow = adj + (size_t)(row0 + i) * N_NODES + jb;
  const unsigned short* bcol = WhT + (size_t)colbase * N_NODES + kbB;

  int buf = 0;
  for (int j = 0; j < N_NODES; j += JT, buf ^= 1) {
    // keep the HBM stream ahead of the barrier-structured loop
    if (j + 2 * JT <= N_NODES)
      __builtin_prefetch(adjrow + j + 2 * JT, 0, 0);

    // ---- scores: p = adj ? exp(leakyrelu(wh1+wh2) - shift) : 0
    i32x4 av = __builtin_nontemporal_load((const i32x4*)(adjrow + j));
    f32x4 w2 = *(const f32x4*)(Wh2 + j + jb);
    float e0 = wh1 + w2.x; e0 = (e0 > 0.f) ? e0 : 0.2f * e0;
    float e1 = wh1 + w2.y; e1 = (e1 > 0.f) ? e1 : 0.2f * e1;
    float e2 = wh1 + w2.z; e2 = (e2 > 0.f) ? e2 : 0.2f * e2;
    float e3 = wh1 + w2.w; e3 = (e3 > 0.f) ? e3 : 0.2f * e3;
    float p0 = (av.x > 0) ? __expf(e0 - shift) : 0.f;
    float p1 = (av.y > 0) ? __expf(e1 - shift) : 0.f;
    float p2 = (av.z > 0) ? __expf(e2 - shift) : 0.f;
    float p3 = (av.w > 0) ? __expf(e3 - shift) : 0.f;
    rsum += (p0 + p1) + (p2 + p3);
    u16x4 pb; pb.x = f2bf(p0); pb.y = f2bf(p1); pb.z = f2bf(p2); pb.w = f2bf(p3);
    *(u16x4*)(&Pt[buf][i * PSTR + jb]) = pb;
    __syncthreads();

    // ---- A fragments: 2 row-groups x 2 K-subtiles from the LDS P tile
    const unsigned short* pt = &Pt[buf][0];
    const unsigned short* pr0 = pt + (0  + arow) * PSTR + kbA;   // rows 0-15
    const unsigned short* pr1 = pt + (16 + arow) * PSTR + kbA;   // rows 16-31
    Frag a00, a01, a10, a11;
    a00.q[0] = *(const u32x4*)(pr0);      a00.q[1] = *(const u32x4*)(pr0 + 16);
    a01.q[0] = *(const u32x4*)(pr0 + 32); a01.q[1] = *(const u32x4*)(pr0 + 48);
    a10.q[0] = *(const u32x4*)(pr1);      a10.q[1] = *(const u32x4*)(pr1 + 16);
    a11.q[0] = *(const u32x4*)(pr1 + 32); a11.q[1] = *(const u32x4*)(pr1 + 48);

    // ---- B fragments (32x16 bf16 per K-subtile), each reused by 2 WMMAs
    const unsigned short* pb0 = bcol + j;         // K-subtile 0
    const unsigned short* pb1 = bcol + j + 32;    // K-subtile 1
    Frag b0, b1;
    b0.q[0] = *(const u32x4*)pb0; b0.q[1] = *(const u32x4*)(pb0 + 16);
    b1.q[0] = *(const u32x4*)pb1; b1.q[1] = *(const u32x4*)(pb1 + 16);

    acc0 = __builtin_amdgcn_wmma_f32_16x16x32_bf16(false, a00.v, false, b0.v,
                                                   (short)0, acc0, false, false);
    acc1 = __builtin_amdgcn_wmma_f32_16x16x32_bf16(false, a10.v, false, b0.v,
                                                   (short)0, acc1, false, false);
    acc0 = __builtin_amdgcn_wmma_f32_16x16x32_bf16(false, a01.v, false, b1.v,
                                                   (short)0, acc0, false, false);
    acc1 = __builtin_amdgcn_wmma_f32_16x16x32_bf16(false, a11.v, false, b1.v,
                                                   (short)0, acc1, false, false);
    // no second barrier: double-buffered P tile
  }

  // ---- row-sum reduction l_i
  lpart[i * 16 + (tid & 15)] = rsum;
  __syncthreads();
  if (tid < ROWS) {
    float s = 0.f;
    for (int q = 0; q < 16; ++q) s += lpart[tid * 16 + q];
    lrow[tid] = s;
  }
  __syncthreads();

  // ---- normalize & store: VGPR r -> row r (lanes 0-15) / r+8 (lanes 16-31)
  const int rofs = (lane < 16) ? 0 : 8;
  for (int r = 0; r < 8; ++r) {
    const int rr = r + rofs;
    const float inv0 = 1.0f / lrow[rr];
    const float inv1 = 1.0f / lrow[16 + rr];
    out[(size_t)(row0 + rr)      * D_OUT + colbase] = acc0[r] * inv0;
    out[(size_t)(row0 + 16 + rr) * D_OUT + colbase] = acc1[r] * inv1;
  }
}

// ---------------------------------------------------------------------------
extern "C" void kernel_launch(void* const* d_in, const int* in_sizes, int n_in,
                              void* d_out, int out_size, void* d_ws, size_t ws_size,
                              hipStream_t stream) {
  const float* X   = (const float*)d_in[0];   // [8192,512]
  const int*   adj = (const int*)  d_in[1];   // [8192,8192]
  const float* W   = (const float*)d_in[2];   // [512,256]
  const float* a   = (const float*)d_in[3];   // [512,1]
  float* out = (float*)d_out;                 // [8192,256]

  // workspace layout (~12.1 MB)
  char* ws = (char*)d_ws;
  float*          Wh  = (float*)ws;                                       // 8 MB
  unsigned short* WhT = (unsigned short*)(ws + (size_t)8  * 1024 * 1024); // 4 MB
  float*          Wh1 = (float*)(ws + (size_t)12 * 1024 * 1024);
  float*          Wh2 = Wh1 + N_NODES;
  float*          sh  = Wh2 + N_NODES;

  k_wh_gemm<<<dim3(N_NODES / 16, D_OUT / 64), 128, 0, stream>>>(X, W, Wh);
  k_transpose_bf16<<<dim3(D_OUT / 32, N_NODES / 32), dim3(32, 8), 0, stream>>>(Wh, WhT);
  k_wh12<<<N_NODES / 8, 256, 0, stream>>>(Wh, a, Wh1, Wh2);
  k_shift<<<1, 256, 0, stream>>>(Wh1, Wh2, sh);
  k_gat_attn<<<N_NODES / ROWS, 512, 0, stream>>>(adj, Wh1, Wh2, WhT, sh, out);
}
